// DAGCN_7705171329819
// MI455X (gfx1250) — matrix operand order repaired and checked by reading, not compiled, not run on hardware
//
#include <hip/hip_runtime.h>
#include <hip/hip_bf16.h>

// Problem constants (from reference)
#define BB 32
#define TT 12
#define NN 512
#define EE 10
#define KK 3
#define CI 64
#define CO 64

typedef __attribute__((ext_vector_type(16))) __bf16       v16bf;
typedef __attribute__((ext_vector_type(8)))  float        v8f;
typedef __attribute__((ext_vector_type(4)))  float        f32x4;
typedef __attribute__((ext_vector_type(4)))  unsigned int u32x4;
typedef __attribute__((ext_vector_type(4)))  int          v4i;

#define AS1 __attribute__((address_space(1)))
#define AS3 __attribute__((address_space(3)))

union BF16x16 {
    v16bf  v;
    __bf16 e[16];
    u32x4  q[2];
};

static __device__ __forceinline__ v8f wmma_bf16(v16bf a, v16bf b, v8f c) {
    // D = A(16x32 bf16) * B(32x16 bf16) + C(16x16 f32)
    return __builtin_amdgcn_wmma_f32_16x16x32_bf16(
        /*neg_a=*/false, a, /*neg_b=*/false, b,
        /*c_mod=*/(short)0, c, /*reuse_a=*/false, /*reuse_b=*/false);
}

// Async global->LDS DMA (CDNA5): 16B per lane, tracked by ASYNCcnt.
#if defined(__gfx1250__) && __has_builtin(__builtin_amdgcn_global_load_async_to_lds_b128) && __has_builtin(__builtin_amdgcn_s_wait_asynccnt)
#define USE_ASYNC_LDS 1
#else
#define USE_ASYNC_LDS 0
#endif

#if USE_ASYNC_LDS
static __device__ __forceinline__ void async_copy16(const void* g, void* lds) {
    // Generic LDS pointer's low 32 bits are the DS byte offset (ISA 10.2);
    // global generic addresses are the flat VA, so int-casts give AS1/AS3 ptrs.
    // Builtin signature (from compiler diagnostic): arg0 = AS1 v4i*, non-const.
    __builtin_amdgcn_global_load_async_to_lds_b128(
        (AS1 v4i*)(uintptr_t)g,
        (AS3 v4i*)(uintptr_t)lds,
        /*imm offset=*/0, /*cpol=*/0);
}
#endif

// ---------------------------------------------------------------------------
// Kernel 1: supports[t,n,:] = softmax(relu(emb[t,n,:] . emb[t,:,:]^T)) (bf16)
//           diag[t,n] = supports[t,n,n] (f32)
// One block per (t,n); 256 threads, each handles 2 columns of the 512-row.
// ---------------------------------------------------------------------------
__global__ __launch_bounds__(256) void adj_kernel(const float* __restrict__ emb,
                                                  __bf16* __restrict__ sup,
                                                  float* __restrict__ diag) {
    const int t = blockIdx.y, n = blockIdx.x;
    const int tid = threadIdx.x;
    const float* ebase = emb + (size_t)t * NN * EE;

    float en[EE];
#pragma unroll
    for (int d = 0; d < EE; ++d) en[d] = ebase[n * EE + d];

    float s[2];
#pragma unroll
    for (int h = 0; h < 2; ++h) {
        const int m = tid + h * 256;
        const float* em = ebase + m * EE;
        float acc = 0.f;
#pragma unroll
        for (int d = 0; d < EE; ++d) acc += en[d] * em[d];
        s[h] = fmaxf(acc, 0.f);  // relu
    }

    __shared__ float red[256];
    // row max
    red[tid] = fmaxf(s[0], s[1]);
    __syncthreads();
    for (int off = 128; off > 0; off >>= 1) {
        if (tid < off) red[tid] = fmaxf(red[tid], red[tid + off]);
        __syncthreads();
    }
    const float rmax = red[0];
    __syncthreads();
    // exp + row sum
    float p0 = __expf(s[0] - rmax);
    float p1 = __expf(s[1] - rmax);
    red[tid] = p0 + p1;
    __syncthreads();
    for (int off = 128; off > 0; off >>= 1) {
        if (tid < off) red[tid] += red[tid + off];
        __syncthreads();
    }
    const float inv = 1.0f / red[0];

    __bf16* srow = sup + ((size_t)t * NN + n) * NN;
    {
        const float v0 = p0 * inv;
        const float v1 = p1 * inv;
        srow[tid]       = (__bf16)v0;
        srow[tid + 256] = (__bf16)v1;
        if (tid == n)        diag[t * NN + n] = v0;
        if (tid + 256 == n)  diag[t * NN + n] = v1;
    }
}

// ---------------------------------------------------------------------------
// Kernel 2: x_g1[t,n,b,c] = sum_m supports[t,n,m] * x[b,t,m,c]   (bf16 out)
// Per block: one (t, b, 64-row n-tile); C tile 64(n) x 64(c); K=512 in 32s.
// 4 waves; wave w owns rows [w*16, w*16+16). LDS: A row-major, B col-major,
// both shaped so each lane's WMMA fragment = two contiguous 16B chunks.
// A tile is staged with async global->LDS DMA when available.
// ---------------------------------------------------------------------------
__global__ __launch_bounds__(128) void gemm1_kernel(const __bf16* __restrict__ sup,
                                                    const float* __restrict__ x,
                                                    __bf16* __restrict__ xg1) {
    const int t = blockIdx.z, b = blockIdx.y, n0 = blockIdx.x * 64;
    const int tid = threadIdx.x;
    const int lane = tid & 31, wave = tid >> 5;
    const int g = lane >> 4, l15 = lane & 15;

    __shared__ __bf16 Als[64 * 32];  // [row r][k]   row-major
    __shared__ __bf16 Bls[64 * 32];  // [col c][k]   col-major

    v8f acc[4];
#pragma unroll
    for (int ct = 0; ct < 4; ++ct)
#pragma unroll
        for (int r = 0; r < 8; ++r) acc[ct][r] = 0.f;

    const __bf16* supt = sup + ((size_t)t * NN + n0) * NN;
    const float*  xt   = x + ((size_t)b * TT + t) * NN * CI;

    // A-tile copy geometry: 256 x 16B chunks, 2 per thread (straight-line).
    const int ar  = tid >> 2;            // rows 0..31 (chunk 0), +32 (chunk 1)
    const int ak8 = (tid & 3) * 8;

    for (int kc = 0; kc < NN; kc += 32) {
        // ---- A tile: 64x32 bf16 (pure copy) ----
#if USE_ASYNC_LDS
        async_copy16(&supt[(size_t)ar * NN + kc + ak8],        &Als[ar * 32 + ak8]);
        async_copy16(&supt[(size_t)(ar + 32) * NN + kc + ak8], &Als[(ar + 32) * 32 + ak8]);
#else
        *(u32x4*)&Als[ar * 32 + ak8] =
            *(const u32x4*)&supt[(size_t)ar * NN + kc + ak8];
        *(u32x4*)&Als[(ar + 32) * 32 + ak8] =
            *(const u32x4*)&supt[(size_t)(ar + 32) * NN + kc + ak8];
#endif
        // ---- B tile: load x rows (f32x4), convert, store col-major bf16 ----
#pragma unroll
        for (int q = 0; q < 4; ++q) {
            const int idx = q * 128 + tid;
            const int k = idx >> 4, c4 = (idx & 15) * 4;
            const f32x4 v = *(const f32x4*)&xt[(size_t)(kc + k) * CI + c4];
#pragma unroll
            for (int w = 0; w < 4; ++w) Bls[(c4 + w) * 32 + k] = (__bf16)v[w];
        }
        if (kc + 32 < NN) {
            // hint next K-chunk of the strided x walk into cache
            __builtin_prefetch(&xt[(size_t)(kc + 32) * CI + (tid & 31) * 8], 0, 1);
        }
#if USE_ASYNC_LDS
        __builtin_amdgcn_s_wait_asynccnt(0);
#endif
        __syncthreads();

        // A fragment (ISA layout): lane row = wave*16 + l15,
        // K = {g*8..g*8+7} U {16+g*8..16+g*8+7}
        BF16x16 af;
        const __bf16* arow = &Als[(wave * 16 + l15) * 32];
        af.q[0] = *(const u32x4*)(arow + g * 8);
        af.q[1] = *(const u32x4*)(arow + 16 + g * 8);

#pragma unroll
        for (int ct = 0; ct < 4; ++ct) {
            // B fragment: lane col = ct*16 + l15, K = {g*8..g*8+7} U {16+g*8..16+g*8+7}
            BF16x16 bf;
            const __bf16* brow = &Bls[(ct * 16 + l15) * 32];
            bf.q[0] = *(const u32x4*)(brow + g * 8);
            bf.q[1] = *(const u32x4*)(brow + 16 + g * 8);
            acc[ct] = wmma_bf16(af.v, bf.v, acc[ct]);
        }
        __syncthreads();
    }

    // store: C/D layout -> row = r + 8*g, col = l15 within 16x16 tile
#pragma unroll
    for (int ct = 0; ct < 4; ++ct) {
        const int c = ct * 16 + l15;
#pragma unroll
        for (int r = 0; r < 8; ++r) {
            const int nrow = n0 + wave * 16 + r + 8 * g;
            xg1[(((size_t)t * NN + nrow) * BB + b) * CI + c] = (__bf16)acc[ct][r];
        }
    }
}

// ---------------------------------------------------------------------------
// Kernel 3: out[b,t,n,o] = relu( X_tn @ Wx + G_tn @ Wg + bias )
//   Wx = sum_d e_d (Wp[d,0] - Wp[d,2]);  Wg = sum_d e_d (Wp[d,1] + 2*diag*Wp[d,2])
// One block per (t,n); 4 waves = 4 o-tiles of 16. M=32(b) x N=64(o) x K=64(i).
// Weight B-fragments built in registers in contiguous-k order (k = 16*g + j), same as A.
// ---------------------------------------------------------------------------
__global__ __launch_bounds__(128) void final_kernel(const float* __restrict__ x,
                                                    const float* __restrict__ emb,
                                                    const float* __restrict__ wp,
                                                    const float* __restrict__ bp,
                                                    const float* __restrict__ diag,
                                                    const __bf16* __restrict__ xg1,
                                                    float* __restrict__ out) {
    const int t = blockIdx.y, n = blockIdx.x;
    const int tid = threadIdx.x;
    const int lane = tid & 31, wave = tid >> 5;
    const int g = lane >> 4, l15 = lane & 15;
    const int o = wave * 16 + l15;  // this lane's output column

    float ev[EE];
    const float* et = emb + ((size_t)t * NN + n) * EE;
#pragma unroll
    for (int d = 0; d < EE; ++d) ev[d] = et[d];
    const float dg2 = 2.0f * diag[t * NN + n];

    // bias[t,n,o] = sum_d e_d * bias_pool[t,d,o]
    float biasv = 0.f;
#pragma unroll
    for (int d = 0; d < EE; ++d) biasv += ev[d] * bp[(t * EE + d) * CO + o];

    // Build weight B-fragments: lane holds K = kt*32 + g*16 + j, col = o
    const float* wpt = wp + (size_t)t * EE * KK * CI * CO;
    BF16x16 wx[2], wg[2];
#pragma unroll
    for (int kt = 0; kt < 2; ++kt) {
        for (int j = 0; j < 16; ++j) {
            const int i = kt * 32 + g * 16 + j;
            const float* p = wpt + (size_t)i * CO + o;  // (d=0,k=0)
            float th0 = 0.f, th1 = 0.f, th2 = 0.f;
#pragma unroll
            for (int d = 0; d < EE; ++d) {
                th0 += ev[d] * p[0];
                th1 += ev[d] * p[CI * CO];
                th2 += ev[d] * p[2 * CI * CO];
                p += KK * CI * CO;
            }
            wx[kt].e[j] = (__bf16)(th0 - th2);
            wg[kt].e[j] = (__bf16)(th1 + dg2 * th2);
        }
    }

    // Accumulators seeded with bias (bias depends only on col -> broadcast)
    v8f acc[2];
#pragma unroll
    for (int mt = 0; mt < 2; ++mt)
#pragma unroll
        for (int r = 0; r < 8; ++r) acc[mt][r] = biasv;

#pragma unroll
    for (int mt = 0; mt < 2; ++mt) {
        const int brow = mt * 16 + l15;  // A-matrix row = batch index
        const float*  xrow = x   + (((size_t)brow * TT + t) * NN + n) * CI;
        const __bf16* grow = xg1 + (((size_t)t * NN + n) * BB + brow) * CI;
#pragma unroll
        for (int kt = 0; kt < 2; ++kt) {
            const int base1 = kt * 32 + g * 16;
            const int base2 = base1 + 8;
            BF16x16 ax, ag;
            const f32x4 x0 = *(const f32x4*)(xrow + base1);
            const f32x4 x1 = *(const f32x4*)(xrow + base1 + 4);
            const f32x4 x2 = *(const f32x4*)(xrow + base2);
            const f32x4 x3 = *(const f32x4*)(xrow + base2 + 4);
#pragma unroll
            for (int j = 0; j < 4; ++j) {
                ax.e[j]      = (__bf16)x0[j];
                ax.e[4 + j]  = (__bf16)x1[j];
                ax.e[8 + j]  = (__bf16)x2[j];
                ax.e[12 + j] = (__bf16)x3[j];
            }
            ag.q[0] = *(const u32x4*)(grow + base1);
            ag.q[1] = *(const u32x4*)(grow + base2);

            acc[mt] = wmma_bf16(ax.v, wx[kt].v, acc[mt]);
            acc[mt] = wmma_bf16(ag.v, wg[kt].v, acc[mt]);
        }
    }

    // relu + store (C/D layout: row = mt*16 + r + 8*g, col = o)
#pragma unroll
    for (int mt = 0; mt < 2; ++mt)
#pragma unroll
        for (int r = 0; r < 8; ++r) {
            const int bb = mt * 16 + r + 8 * g;
            out[(((size_t)bb * TT + t) * NN + n) * CO + o] = fmaxf(acc[mt][r], 0.f);
        }
}

// ---------------------------------------------------------------------------
extern "C" void kernel_launch(void* const* d_in, const int* in_sizes, int n_in,
                              void* d_out, int out_size, void* d_ws, size_t ws_size,
                              hipStream_t stream) {
    (void)in_sizes; (void)n_in; (void)out_size; (void)ws_size;
    const float* x   = (const float*)d_in[0];  // [B,T,N,CI]
    const float* emb = (const float*)d_in[1];  // [T,N,E]
    const float* wp  = (const float*)d_in[2];  // [T,E,K,CI,CO]
    const float* bp  = (const float*)d_in[3];  // [T,E,CO]
    float* out = (float*)d_out;                // [B,T,N,CO]

    // Workspace layout (~31.5 MB total)
    char* ws = (char*)d_ws;
    __bf16* sup = (__bf16*)ws;                                // [T,N,N] bf16
    const size_t sup_bytes = (size_t)TT * NN * NN * 2;
    float* diag = (float*)(ws + sup_bytes);                   // [T,N] f32
    const size_t diag_bytes = (size_t)TT * NN * 4;
    __bf16* xg1 = (__bf16*)(ws + sup_bytes + diag_bytes);     // [T,N,B,CI] bf16

    adj_kernel<<<dim3(NN, TT), 256, 0, stream>>>(emb, sup, diag);
    gemm1_kernel<<<dim3(NN / 64, BB, TT), 128, 0, stream>>>(sup, x, xg1);
    final_kernel<<<dim3(NN, TT), 128, 0, stream>>>(x, emb, wp, bp, diag, xg1, out);
}
